// QuantizedPermutationLinear_6459630813324
// MI455X (gfx1250) — 8-bit matrix operand order repaired and checked by reading, compiled, not run on hardware
//
#include <hip/hip_runtime.h>
#include <math.h>


#define B_      4
#define S_      2048
#define FDIM    4096
#define MROWS   (B_ * S_)
#define ROUNDS  12
#define NBLK    (FDIM / 2)

typedef int      v8i __attribute__((ext_vector_type(8)));
typedef int      v4i __attribute__((ext_vector_type(4)));
typedef unsigned v4u __attribute__((ext_vector_type(4)));

#if defined(__has_builtin)
#if __has_builtin(__builtin_amdgcn_tensor_load_to_lds)
#define HAVE_TDM 1
#endif
#endif
#ifndef HAVE_TDM
#define HAVE_TDM 0
#endif

__device__ __forceinline__ unsigned fenc(float f) {
  unsigned u = __float_as_uint(f);
  return (u & 0x80000000u) ? ~u : (u | 0x80000000u);
}
__device__ __forceinline__ float fdec(unsigned u) {
  unsigned b = (u & 0x80000000u) ? (u ^ 0x80000000u) : ~u;
  return __uint_as_float(b);
}

__device__ __forceinline__ void rotate_row(float* row,
                                           const int* __restrict__ perms,
                                           const float* __restrict__ mats,
                                           int tid) {
  for (int r = 0; r < ROUNDS; ++r) {
    const int*    p = perms + r * FDIM;
    const float4* m = ((const float4*)mats) + r * NBLK;
    float c[16];
#pragma unroll
    for (int i = 0; i < 8; ++i) {
      int k = tid * 8 + i;
      int p0 = p[2 * k];
      int p1 = p[2 * k + 1];
      float a = row[p0];
      float b = row[p1];
      float4 mm = m[k];
      c[2 * i]     = mm.x * a + mm.y * b;
      c[2 * i + 1] = mm.z * a + mm.w * b;
    }
    __syncthreads();
#pragma unroll
    for (int i = 0; i < 8; ++i) {
      row[tid * 16 + 2 * i]     = c[2 * i];
      row[tid * 16 + 2 * i + 1] = c[2 * i + 1];
    }
    __syncthreads();
  }
}

__global__ void init_minmax_kernel(unsigned* mm) {
  int t = threadIdx.x;
  if (t < 2) {
    mm[t * 2 + 0] = 0xFFFFFFFFu;
    mm[t * 2 + 1] = 0u;
  }
}

__global__ void __launch_bounds__(256)
rot_minmax_kernel(const float* __restrict__ src,
                  const int*   __restrict__ perms,
                  const float* __restrict__ mats,
                  unsigned*    __restrict__ mm) {
  __shared__ __align__(16) float row[FDIM];
  __shared__ float red[256];
  const int tid = threadIdx.x;
  const float4* s4 = (const float4*)(src + (size_t)blockIdx.x * FDIM);
  float4* r4 = (float4*)row;
#pragma unroll
  for (int q = 0; q < 4; ++q) r4[q * 256 + tid] = s4[q * 256 + tid];
  __syncthreads();

  rotate_row(row, perms, mats, tid);

  float lmin = 3.402823466e38f, lmax = -3.402823466e38f;
#pragma unroll
  for (int i = 0; i < 16; ++i) {
    float v = row[tid * 16 + i];
    lmin = fminf(lmin, v);
    lmax = fmaxf(lmax, v);
  }
  red[tid] = lmin; __syncthreads();
  for (int s = 128; s > 0; s >>= 1) {
    if (tid < s) red[tid] = fminf(red[tid], red[tid + s]);
    __syncthreads();
  }
  float bmin = red[0]; __syncthreads();
  red[tid] = lmax; __syncthreads();
  for (int s = 128; s > 0; s >>= 1) {
    if (tid < s) red[tid] = fmaxf(red[tid], red[tid + s]);
    __syncthreads();
  }
  if (tid == 0) {
    atomicMin(&mm[0], fenc(bmin));
    atomicMax(&mm[1], fenc(red[0]));
  }
}

__global__ void __launch_bounds__(256)
rot_quant_kernel(const float* __restrict__ src,
                 const int*   __restrict__ perms,
                 const float* __restrict__ mats,
                 const unsigned* __restrict__ mm,
                 unsigned char* __restrict__ dstq,
                 float* __restrict__ rowsum) {
  __shared__ __align__(16) float row[FDIM];
  __shared__ float red[256];
  const int tid = threadIdx.x;
  const float4* s4 = (const float4*)(src + (size_t)blockIdx.x * FDIM);
  float4* r4 = (float4*)row;
#pragma unroll
  for (int q = 0; q < 4; ++q) r4[q * 256 + tid] = s4[q * 256 + tid];
  __syncthreads();

  rotate_row(row, perms, mats, tid);

  const float lo  = fdec(mm[0]);
  const float hi  = fdec(mm[1]);
  const float inv = 255.0f / (hi - lo);

  unsigned pk[4];
  float lsum = 0.0f;
#pragma unroll
  for (int w = 0; w < 4; ++w) {
    unsigned packed = 0;
#pragma unroll
    for (int b = 0; b < 4; ++b) {
      float v = row[tid * 16 + w * 4 + b];
      float q = fminf(fmaxf(rintf((v - lo) * inv), 0.0f), 255.0f);
      lsum += q;
      packed |= ((unsigned)q) << (8 * b);
    }
    pk[w] = packed;
  }
  uint4 p4; p4.x = pk[0]; p4.y = pk[1]; p4.z = pk[2]; p4.w = pk[3];
  ((uint4*)(dstq + (size_t)blockIdx.x * FDIM))[tid] = p4;

  red[tid] = lsum; __syncthreads();
  for (int s = 128; s > 0; s >>= 1) {
    if (tid < s) red[tid] += red[tid + s];
    __syncthreads();
  }
  if (tid == 0) rowsum[blockIdx.x] = red[0];
}

__device__ __forceinline__ void tdm_load_2d(unsigned lds_off, const void* gptr,
                                            unsigned tile_rows, unsigned tile_bytes,
                                            unsigned row_stride, unsigned tensor_rows) {
#if HAVE_TDM
  unsigned long long ga = (unsigned long long)(size_t)gptr;
  v4u g0;
  g0.x = 1u;
  g0.y = lds_off;
  g0.z = (unsigned)ga;
  g0.w = (unsigned)((ga >> 32) & 0x01FFFFFFull) | (2u << 30);
  v8i g1;
  g1[0] = 0;
  g1[1] = (int)((row_stride & 0xFFFFu) << 16);
  g1[2] = (int)((row_stride >> 16) | ((tensor_rows & 0xFFFFu) << 16));
  g1[3] = (int)((tensor_rows >> 16) | (tile_bytes << 16));
  g1[4] = (int)tile_rows;
  g1[5] = (int)row_stride;
  g1[6] = 0;
  g1[7] = 0;
  v4i z4 = {0, 0, 0, 0};
#if defined(__clang_major__) && (__clang_major__ >= 23)
  v8i z8 = {0, 0, 0, 0, 0, 0, 0, 0};
  __builtin_amdgcn_tensor_load_to_lds(g0, g1, z4, z4, z8, 0);
#else
  __builtin_amdgcn_tensor_load_to_lds(g0, g1, z4, z4, 0);
#endif
#else
  (void)lds_off; (void)gptr; (void)tile_rows; (void)tile_bytes;
  (void)row_stride; (void)tensor_rows;
#endif
}


struct Slab { uint4 a0, a1, b0, b1, b2, b3; };

__device__ __forceinline__ Slab load_slab(const unsigned char* aptr,
                                          const unsigned char* bptr, int kk) {
  Slab s;
  s.a0 = *(const uint4*)(aptr + kk);
  s.a1 = *(const uint4*)(aptr + kk + 16);
  s.b0 = *(const uint4*)(bptr + kk);
  s.b1 = *(const uint4*)(bptr + kk + 16);
  s.b2 = *(const uint4*)(bptr + kk + 32);
  s.b3 = *(const uint4*)(bptr + kk + 48);
  return s;
}

__device__ __forceinline__ void store_slab(unsigned (*As)[16], unsigned (*Bs)[16],
                                           const Slab& s, int arow, int acol, int brow) {
  *(uint4*)&As[arow][acol]     = s.a0;
  *(uint4*)&As[arow][acol + 4] = s.a1;
  *(uint4*)&Bs[brow][0]        = s.b0;
  *(uint4*)&Bs[brow][4]        = s.b1;
  *(uint4*)&Bs[brow][8]        = s.b2;
  *(uint4*)&Bs[brow][12]       = s.b3;
}

__device__ __forceinline__ void mma_slab(const unsigned (*As)[16], const unsigned (*Bs)[16],
                                         v8i acc[4][4], int wm, int wn, int l16, int hi) {
  v8i b[4];
#pragma unroll
  for (int j = 0; j < 4; ++j) {
    const int rowB = wn * 64 + j * 16 + l16;
#pragma unroll
    for (int v = 0; v < 8; ++v) {
      const int dw = ((v >> 1) << 2) + (v & 1) + (hi << 1);
      b[j][v] = (int)Bs[rowB][dw];
    }
  }
#pragma unroll
  for (int i = 0; i < 4; ++i) {
    v8i a;
    const int rowA = wm * 64 + i * 16 + l16;
#pragma unroll
    for (int v = 0; v < 8; ++v) {
      const int dw = ((v >> 1) << 2) + (v & 1) + (hi << 1);
      a[v] = (int)As[rowA][dw];
    }
#pragma unroll
    for (int j = 0; j < 4; ++j)
      acc[i][j] = __builtin_amdgcn_wmma_i32_16x16x64_iu8(
           false, a,  false, b[j], acc[i][j],
           false,  false);
  }
}

__global__ void __launch_bounds__(256)
gemm_iu8_kernel(const unsigned char* __restrict__ Xq,
                const unsigned char* __restrict__ Wq,
                const float* __restrict__ xsum,
                const float* __restrict__ wsum,
                const float* __restrict__ bias,
                const unsigned* __restrict__ mm,
                float* __restrict__ out) {
  __shared__ __align__(16) unsigned As[2][128][16];
  __shared__ __align__(16) unsigned Bs[2][256][16];

  const int tid  = threadIdx.x;
  const int lane = tid & 31;
  const int wv   = tid >> 5;
  const int wm   = wv >> 2;
  const int wn   = wv & 3;
  const int hi   = lane >> 4;
  const int l16  = lane & 15;
  const int m0   = blockIdx.y * 128;
  const int n0   = blockIdx.x * 256;

  v8i acc[4][4];
#pragma unroll
  for (int i = 0; i < 4; ++i)
#pragma unroll
    for (int j = 0; j < 4; ++j)
      acc[i][j] = (v8i){0, 0, 0, 0, 0, 0, 0, 0};

#if HAVE_TDM
  const unsigned ldsA[2] = { (unsigned)(size_t)&As[0][0][0],
                             (unsigned)(size_t)&As[1][0][0] };
  const unsigned ldsB[2] = { (unsigned)(size_t)&Bs[0][0][0],
                             (unsigned)(size_t)&Bs[1][0][0] };
  const unsigned char* gA = Xq + (size_t)m0 * FDIM;
  const unsigned char* gB = Wq + (size_t)n0 * FDIM;
  const bool issuer = (__builtin_amdgcn_readfirstlane(wv) == 0);

  if (issuer) {
    tdm_load_2d(ldsA[0], gA, 128, 64, FDIM, 128);
    tdm_load_2d(ldsB[0], gB, 256, 64, FDIM, 256);
    __builtin_amdgcn_s_wait_tensorcnt(0);
  }
  __syncthreads();

  for (int kt = 0; kt < 63; ++kt) {
    const int cur = kt & 1;
    if (issuer) {
      tdm_load_2d(ldsA[cur ^ 1], gA + (kt + 1) * 64, 128, 64, FDIM, 128);
      tdm_load_2d(ldsB[cur ^ 1], gB + (kt + 1) * 64, 256, 64, FDIM, 256);
    }
    mma_slab(As[cur], Bs[cur], acc, wm, wn, l16, hi);
    if (issuer) __builtin_amdgcn_s_wait_tensorcnt(0);
    __syncthreads();
  }
  mma_slab(As[1], Bs[1], acc, wm, wn, l16, hi);
#else
  const int arow = tid >> 1;
  const int acol = (tid & 1) * 8;
  const int brow = tid;
  const unsigned char* aptr = Xq + (size_t)(m0 + arow) * FDIM + (size_t)acol * 4;
  const unsigned char* bptr = Wq + (size_t)(n0 + brow) * FDIM;

  Slab s = load_slab(aptr, bptr, 0);
  store_slab(As[0], Bs[0], s, arow, acol, brow);
  __syncthreads();

  for (int kt = 0; kt < 63; ++kt) {
    const int cur = kt & 1;
    s = load_slab(aptr, bptr, (kt + 1) * 64);
    __builtin_prefetch(aptr + (kt + 2) * 64, 0, 1);
    __builtin_prefetch(bptr + (kt + 2) * 64, 0, 1);
    mma_slab(As[cur], Bs[cur], acc, wm, wn, l16, hi);
    store_slab(As[cur ^ 1], Bs[cur ^ 1], s, arow, acol, brow);
    __syncthreads();
  }
  mma_slab(As[1], Bs[1], acc, wm, wn, l16, hi);
#endif

  const float lx = fdec(mm[0]), hx = fdec(mm[1]);
  const float lw = fdec(mm[2]), hw = fdec(mm[3]);
  const float sx = (hx - lx) * (1.0f / 255.0f);
  const float sw = (hw - lw) * (1.0f / 255.0f);
  const float ss  = sx * sw;
  const float cst = (float)FDIM * lx * lw;

#pragma unroll
  for (int i = 0; i < 4; ++i) {
#pragma unroll
    for (int j = 0; j < 4; ++j) {
      const int n = n0 + wn * 64 + j * 16 + l16;
      const float ncorr = lx * sw * wsum[n] + cst + bias[n];
#pragma unroll
      for (int e = 0; e < 8; ++e) {
        const int m = m0 + wm * 64 + i * 16 + e + hi * 8;
        out[(size_t)m * FDIM + n] =
            ss * (float)acc[i][j][e] + sx * lw * xsum[m] + ncorr;
      }
    }
  }
}

extern "C" void kernel_launch(void* const* d_in, const int* in_sizes, int n_in,
                              void* d_out, int out_size, void* d_ws, size_t ws_size,
                              hipStream_t stream) {
  (void)in_sizes; (void)n_in; (void)out_size; (void)ws_size;

  const float* x      = (const float*)d_in[0];
  const float* weight = (const float*)d_in[1];
  const float* bias   = (const float*)d_in[2];
  const float* mats   = (const float*)d_in[3];
  const int*   perms  = (const int*)  d_in[4];
  float* out = (float*)d_out;

  unsigned char* ws = (unsigned char*)d_ws;
  unsigned char* xq = ws;
  unsigned char* wq = ws + (size_t)33554432;
  float* xsum  = (float*)(ws + (size_t)50331648);
  float* wsum  = (float*)(ws + (size_t)50364416);
  unsigned* mm = (unsigned*)(ws + (size_t)50380800);

  init_minmax_kernel<<<1, 64, 0, stream>>>(mm);
  rot_minmax_kernel<<<MROWS, 256, 0, stream>>>(x,      perms, mats, mm + 0);
  rot_minmax_kernel<<<FDIM,  256, 0, stream>>>(weight, perms, mats, mm + 2);
  rot_quant_kernel<<<MROWS, 256, 0, stream>>>(x,      perms, mats, mm + 0, xq, xsum);
  rot_quant_kernel<<<FDIM,  256, 0, stream>>>(weight, perms, mats, mm + 2, wq, wsum);

  dim3 grid(FDIM / 256, MROWS / 128);
  gemm_iu8_kernel<<<grid, 256, 0, stream>>>(xq, wq, xsum, wsum, bias, mm, out);
}
